// Model_60971355734628
// MI455X (gfx1250) — 16-bit and 8-bit matrix operand order repaired and checked by reading, compiled, not run on hardware
//
#include <hip/hip_runtime.h>
#include <math.h>

#define HIDDEN 256
#define SEQ    999
#define GATES  1024
#define MPAD   1008
#define FLATN  (SEQ*HIDDEN)

#define WSCALE 64.0f
#define HSCALE 16.0f
#define INVSC  (1.0f / (64.0f * 16.0f))

typedef __bf16 bf16_t;
typedef __attribute__((ext_vector_type(16))) __bf16 v16bf;
typedef __attribute__((ext_vector_type(8)))  __bf16 v8bf;
typedef __attribute__((ext_vector_type(8)))  float  v8f;
typedef __attribute__((ext_vector_type(4)))  float  v4f;
typedef __attribute__((ext_vector_type(8)))  int    v8i;
typedef __attribute__((ext_vector_type(4)))  int    v4i;

union V16 { v16bf v; v8bf h[2]; };
union V8I { v8i v; unsigned long long q[4]; v4i x[2]; };

__device__ __forceinline__ unsigned char f32_to_fp8(float x) {
    unsigned u = __float_as_uint(x);
    unsigned s = (u >> 24) & 0x80u;
    unsigned be = (u >> 23) & 0xFFu;
    unsigned man = u & 0x7FFFFFu;
    if (be == 0) return (unsigned char)s;
    int E = (int)be - 127 + 7;
    if (E >= 16) return (unsigned char)(s | 0x7Eu);
    if (E >= 1) {
        unsigned m3 = man >> 20;
        m3 += (man >> 19) & 1u;
        if (m3 > 7u) { m3 = 0u; ++E;
                       if (E >= 16) return (unsigned char)(s | 0x7Eu); }
        return (unsigned char)(s | ((unsigned)E << 3) | m3);
    }
    int sh = 1 - E;
    if (sh > 3) return (unsigned char)s;
    unsigned full = 0x800000u | man;
    unsigned m3 = full >> (20 + sh);
    m3 += (full >> (19 + sh)) & 1u;
    if (m3 > 7u) return (unsigned char)(s | 0x08u);
    return (unsigned char)(s | m3);
}

__global__ void __launch_bounds__(256)
k_init(const float* __restrict__ Whh0, const float* __restrict__ Whh1,
       const float* __restrict__ Wih1,
       unsigned char* __restrict__ Whh0f8, unsigned char* __restrict__ Whh1f8,
       bf16_t* __restrict__ Wih1b, bf16_t* __restrict__ hs0b,
       float* __restrict__ z1raw) {
    int i = blockIdx.x * 256 + threadIdx.x;
    const int NW = GATES * HIDDEN;
    if (i < NW)            Whh0f8[i]       = f32_to_fp8(Whh0[i] * WSCALE);
    else if (i < 2 * NW)   Whh1f8[i - NW]  = f32_to_fp8(Whh1[i - NW] * WSCALE);
    else if (i < 3 * NW)   Wih1b[i - 2*NW] = (bf16_t)Wih1[i - 2*NW];
    else {
        int j = i - 3 * NW;
        const int PADN = (MPAD - SEQ) * HIDDEN;
        if (j < PADN)             hs0b[SEQ * HIDDEN + j] = (bf16_t)0.f;
        else if (j < PADN + 256)  z1raw[j - PADN] = 0.f;
    }
}

__global__ void __launch_bounds__(256)
k_xw0(const float* __restrict__ X, const float* __restrict__ Wih0,
      const float* __restrict__ bih0, const float* __restrict__ bhh0,
      float* __restrict__ xW0) {
    int i = blockIdx.x * 256 + threadIdx.x;
    int t = i / GATES, g = i % GATES;
    xW0[i] = Wih0[2*g] * X[2*t] + Wih0[2*g+1] * X[2*t+1] + bih0[g] + bhh0[g];
}

__global__ void __launch_bounds__(1024)
k_lstm_scan(const float* __restrict__ xW, const unsigned char* __restrict__ Wf8,
            bf16_t* __restrict__ hs_b, float* __restrict__ hs_f) {
    __shared__ unsigned char h8[HIDDEN] __attribute__((aligned(16)));
    __shared__ float gates[GATES];

    const int tid  = threadIdx.x;
    const int lane = tid & 31;
    const int wave = tid >> 5;
    const int kh   = lane >> 4;
    const int l15  = lane & 15;
    const int koff = kh * 16;
    const int g0   = wave * 16;
    const int g1   = g0 + 512;

    V8I a0[4], a1[4];
    #pragma unroll
    for (int kc = 0; kc < 4; ++kc) {
        const unsigned char* p0 = Wf8 + (size_t)(g0 + l15) * HIDDEN + kc * 64 + koff;
        const unsigned char* p1 = Wf8 + (size_t)(g1 + l15) * HIDDEN + kc * 64 + koff;
        #pragma unroll
        for (int j = 0; j < 4; ++j) {
            a0[kc].q[j] = *(const unsigned long long*)(p0 + (j & 1) * 8 + (j >> 1) * 32);
            a1[kc].q[j] = *(const unsigned long long*)(p1 + (j & 1) * 8 + (j >> 1) * 32);
        }
    }

    float c = 0.f;
    if (tid < HIDDEN) h8[tid] = 0;
    __syncthreads();

    for (int t = 0; t < SEQ; ++t) {
        v8f acc0 = {}; v8f acc1 = {};
        #pragma unroll
        for (int kc = 0; kc < 4; ++kc) {
            const int k0 = kc * 64;
            V8I b;
            b.x[0] = *(const v4i*)(h8 + k0 + kh * 16);
            b.x[1] = *(const v4i*)(h8 + k0 + kh * 16 + 32);
            acc0 = __builtin_amdgcn_wmma_f32_16x16x64_fp8_fp8(
                a0[kc].v, b.v, (short)0, acc0, false, false);
            acc1 = __builtin_amdgcn_wmma_f32_16x16x64_fp8_fp8(
                a1[kc].v, b.v, (short)0, acc1, false, false);
        }
        if (l15 == 0) {
            const float* xwt = xW + (size_t)t * GATES;
            const int mb = kh * 8;
            #pragma unroll
            for (int r = 0; r < 8; ++r) {
                gates[g0 + mb + r] = acc0[r] * INVSC + xwt[g0 + mb + r];
                gates[g1 + mb + r] = acc1[r] * INVSC + xwt[g1 + mb + r];
            }
        }
        __syncthreads();
        if (tid < HIDDEN) {
            float gi = gates[tid];
            float gf = gates[HIDDEN + tid];
            float gg = gates[2 * HIDDEN + tid];
            float go = gates[3 * HIDDEN + tid];
            float si = 1.f / (1.f + __expf(-gi));
            float sf = 1.f / (1.f + __expf(-gf));
            float tg = tanhf(gg);
            float so = 1.f / (1.f + __expf(-go));
            c = sf * c + si * tg;
            float h = so * tanhf(c);
            h8[tid] = f32_to_fp8(h * HSCALE);
            hs_b[(size_t)t * HIDDEN + tid] = (bf16_t)h;
            hs_f[(size_t)t * HIDDEN + tid] = h;
        }
        __syncthreads();
    }
}

__global__ void __launch_bounds__(256)
k_gemm_xw1(const bf16_t* __restrict__ A, const bf16_t* __restrict__ Bw,
           const float* __restrict__ bih, const float* __restrict__ bhh,
           float* __restrict__ xW1) {
    const int tid  = threadIdx.x;
    const int lane = tid & 31;
    const int wave = tid >> 5;
    const int tile = blockIdx.x * 8 + wave;
    if (tile >= 63 * 64) return;
    const int mt = tile >> 6, nt = tile & 63;
    const int kh = lane >> 4, l15 = lane & 15;
    const int tbase = mt * 16, gbase = nt * 16;

    v8f acc = {};
    #pragma unroll
    for (int kc = 0; kc < 8; ++kc) {
        const int k0 = kc * 32;
        V16 a, b;
        const bf16_t* pa = A + (size_t)(tbase + l15) * HIDDEN + k0 + kh * 8;
        a.h[0] = *(const v8bf*)(pa);
        a.h[1] = *(const v8bf*)(pa + 16);
        const bf16_t* pb = Bw + (size_t)(gbase + l15) * HIDDEN + k0 + kh * 8;
        b.h[0] = *(const v8bf*)(pb);
        b.h[1] = *(const v8bf*)(pb + 16);
        acc = __builtin_amdgcn_wmma_f32_16x16x32_bf16(
            false, a.v, false, b.v, (short)0, acc, false, false);
    }
    const int g = gbase + l15;
    const float bias = bih[g] + bhh[g];
    const int mb = kh * 8;
    #pragma unroll
    for (int r = 0; r < 8; ++r) {
        int t = tbase + mb + r;
        if (t < SEQ) xW1[(size_t)t * GATES + g] = acc[r] + bias;
    }
}

__global__ void __launch_bounds__(256)
k_w1_gemv(const float* __restrict__ W1, const float* __restrict__ flat,
          float* __restrict__ z1raw) {
    const int row = blockIdx.y, chunk = blockIdx.x;
    const int Lv = FLATN / 4;
    const int Cv = Lv / 8;
    const v4f* Wr = (const v4f*)(W1 + (size_t)row * FLATN);
    const v4f* Fv = (const v4f*)flat;
    float s = 0.f;
    for (int i = chunk * Cv + threadIdx.x; i < (chunk + 1) * Cv; i += 256) {
        __builtin_prefetch(&Wr[i + 512], 0, 0);
        v4f w = Wr[i], f = Fv[i];
        s += w[0]*f[0] + w[1]*f[1] + w[2]*f[2] + w[3]*f[3];
    }
    __shared__ float red[256];
    red[threadIdx.x] = s;
    __syncthreads();
    for (int st = 128; st > 0; st >>= 1) {
        if (threadIdx.x < st) red[threadIdx.x] += red[threadIdx.x + st];
        __syncthreads();
    }
    if (threadIdx.x == 0) atomicAdd(z1raw + row, red[0]);
}

__device__ __forceinline__ float softplus_f(float x) {
    return x > 20.f ? x : log1pf(__expf(x));
}

__global__ void __launch_bounds__(256)
k_head(const float* __restrict__ z1raw, const float* __restrict__ b1,
       const float* __restrict__ W2, const float* __restrict__ b2,
       const float* __restrict__ W3, const float* __restrict__ b3,
       const float* __restrict__ data, float* __restrict__ out) {
    __shared__ float z1[256];
    __shared__ float z2[128];
    __shared__ float abcd[4];
    const int tid = threadIdx.x;
    z1[tid] = softplus_f(z1raw[tid] + b1[tid]);
    __syncthreads();
    if (tid < 128) {
        float s = b2[tid];
        const float* w = W2 + tid * 256;
        for (int j = 0; j < 256; ++j) s += w[j] * z1[j];
        z2[tid] = softplus_f(s);
    }
    __syncthreads();
    if (tid < 4) {
        float s = b3[tid];
        const float* w = W3 + tid * 128;
        for (int j = 0; j < 128; ++j) s += w[j] * z2[j];
        abcd[tid] = s;
    }
    __syncthreads();
    if (tid == 0) {
        const float a = abcd[0], b = abcd[1], cc = abcd[2], d = abcd[3];
        float R = data[0], J = data[1];
        const float dt = 0.01f;
        for (int t = 0; t < SEQ; ++t) {
            float RJ = R * J;
            float Rn = R + dt * (a * R - b * RJ);
            float Jn = J + dt * (d * RJ - cc * J);
            R = Rn; J = Jn;
            out[2 * t] = R; out[2 * t + 1] = J;
        }
    }
}

extern "C" void kernel_launch(void* const* d_in, const int* in_sizes, int n_in,
                              void* d_out, int out_size, void* d_ws, size_t ws_size,
                              hipStream_t stream) {
    (void)in_sizes; (void)n_in; (void)out_size; (void)ws_size;
    const float* X    = (const float*)d_in[0];
    const float* data = (const float*)d_in[1];
    const float* Wih0 = (const float*)d_in[2];
    const float* Whh0 = (const float*)d_in[3];
    const float* bih0 = (const float*)d_in[4];
    const float* bhh0 = (const float*)d_in[5];
    const float* Wih1 = (const float*)d_in[6];
    const float* Whh1 = (const float*)d_in[7];
    const float* bih1 = (const float*)d_in[8];
    const float* bhh1 = (const float*)d_in[9];
    const float* W1   = (const float*)d_in[10];
    const float* b1   = (const float*)d_in[11];
    const float* W2   = (const float*)d_in[12];
    const float* b2   = (const float*)d_in[13];
    const float* W3   = (const float*)d_in[14];
    const float* b3   = (const float*)d_in[15];

    char* ws = (char*)d_ws;
    size_t off = 0;
    auto alloc = [&](size_t bytes) -> char* {
        char* p = ws + off;
        off += (bytes + 255) & ~(size_t)255;
        return p;
    };
    unsigned char* Whh0f8 = (unsigned char*)alloc((size_t)GATES * HIDDEN);
    unsigned char* Whh1f8 = (unsigned char*)alloc((size_t)GATES * HIDDEN);
    bf16_t* Wih1b = (bf16_t*)alloc((size_t)GATES * HIDDEN * sizeof(bf16_t));
    bf16_t* hs0b  = (bf16_t*)alloc((size_t)MPAD * HIDDEN * sizeof(bf16_t));
    float*  xW0   = (float*)alloc((size_t)SEQ * GATES * sizeof(float));
    float*  xW1   = (float*)alloc((size_t)SEQ * GATES * sizeof(float));
    float*  hs1f  = (float*)alloc((size_t)SEQ * HIDDEN * sizeof(float));
    float*  z1raw = (float*)alloc(256 * sizeof(float));

    k_init<<<3082, 256, 0, stream>>>(Whh0, Whh1, Wih1, Whh0f8, Whh1f8, Wih1b,
                                     hs0b, z1raw);
    k_xw0<<<(SEQ * GATES) / 256, 256, 0, stream>>>(X, Wih0, bih0, bhh0, xW0);
    k_lstm_scan<<<1, 1024, 0, stream>>>(xW0, Whh0f8, hs0b, hs1f);
    k_gemm_xw1<<<(63 * 64) / 8, 256, 0, stream>>>(hs0b, Wih1b, bih1, bhh1, xW1);
    k_lstm_scan<<<1, 1024, 0, stream>>>(xW1, Whh1f8, hs0b, hs1f);
    k_w1_gemv<<<dim3(8, 256), 256, 0, stream>>>(W1, hs1f, z1raw);
    k_head<<<1, 256, 0, stream>>>(z1raw, b1, W2, b2, W3, b3, data,
                                  (float*)d_out);
}
